// Polynormer_18348100288981
// MI455X (gfx1250) — compile-verified, matrix operand order checked by reading, not run on hardware
//
#include <hip/hip_runtime.h>
#include <hip/hip_bf16.h>

// ---------------------------------------------------------------------------
// Polynormer on MI455X (gfx1250): bf16 WMMA GEMMs, async global->LDS staging,
// fused wave32 epilogues. N=100000, C=128, E=1.6M, L=7 local, G=2 global, O=40.
// ---------------------------------------------------------------------------

typedef __attribute__((ext_vector_type(16))) __bf16          v16bf;
typedef __attribute__((ext_vector_type(16))) unsigned short  v16us;
typedef __attribute__((ext_vector_type(8)))  float           v8f;

#define C_DIM 128
#define BETA  0.9f

__device__ __forceinline__ unsigned short f2bf(float f) {
    unsigned u = __float_as_uint(f);
    unsigned r = u + 0x7fffu + ((u >> 16) & 1u);   // round-to-nearest-even
    return (unsigned short)(r >> 16);
}

__device__ __forceinline__ float wred(float v) {
    #pragma unroll
    for (int m = 16; m >= 1; m >>= 1) v += __shfl_xor(v, m, 32);
    return v;
}

// async 16B global -> LDS copy (gfx1250, ASYNCcnt-tracked)
__device__ __forceinline__ void async_g2l_b128(void* lds_dst, const void* gaddr) {
    unsigned lds = (unsigned)(uintptr_t)lds_dst;            // low 32 bits = LDS offset
    unsigned long long ga = (unsigned long long)(uintptr_t)gaddr;
    asm volatile("global_load_async_to_lds_b128 %0, %1, off"
                 :: "v"(lds), "v"(ga) : "memory");
}
__device__ __forceinline__ void async_wait0() {
    asm volatile("s_wait_asynccnt 0x0" ::: "memory");
}

// ---------------- elementwise helpers ----------------

__global__ void k_zero_f32(float* p, size_t n) {
    size_t i = (size_t)blockIdx.x * blockDim.x + threadIdx.x;
    size_t stride = (size_t)gridDim.x * blockDim.x;
    for (; i < n; i += stride) p[i] = 0.0f;
}

__global__ void k_cvt_bf16(const float* __restrict__ s, unsigned short* __restrict__ d, size_t n) {
    size_t i = (size_t)blockIdx.x * blockDim.x + threadIdx.x;
    size_t stride = (size_t)gridDim.x * blockDim.x;
    for (; i < n; i += stride) d[i] = f2bf(s[i]);
}

__global__ void k_deg(const int* __restrict__ eidx, int E, float* __restrict__ deg) {
    int i = blockIdx.x * blockDim.x + threadIdx.x;
    int stride = gridDim.x * blockDim.x;
    for (; i < E; i += stride) atomicAdd(&deg[eidx[E + i]], 1.0f);
}

__global__ void k_dinv(const float* __restrict__ deg, float* __restrict__ dinv, int n) {
    int i = blockIdx.x * blockDim.x + threadIdx.x;
    if (i < n) dinv[i] = rsqrtf(deg[i] + 1.0f);   // self-loop adds 1
}

// agg = xw * dinv^2 (self-loop term) + bg
__global__ void k_scatter_init(const float* __restrict__ xw, const float* __restrict__ dinv,
                               const float* __restrict__ bg, float* __restrict__ agg, size_t total) {
    size_t i = (size_t)blockIdx.x * blockDim.x + threadIdx.x;
    size_t stride = (size_t)gridDim.x * blockDim.x;
    for (; i < total; i += stride) {
        int n = (int)(i >> 7), c = (int)(i & 127);
        float dv = dinv[n];
        agg[i] = xw[i] * dv * dv + bg[c];
    }
}

// one wave per edge, 4 channels per lane: agg[dst] += xw[src] * dinv[src]*dinv[dst]
__global__ __launch_bounds__(256) void k_edge_scatter(const int* __restrict__ eidx, int E,
                                                      const float* __restrict__ dinv,
                                                      const float* __restrict__ xw,
                                                      float* __restrict__ agg) {
    int wave = threadIdx.x >> 5, lane = threadIdx.x & 31;
    int e = blockIdx.x * 8 + wave;
    if (e >= E) return;
    int src = eidx[e], dst = eidx[E + e];
    float w = dinv[src] * dinv[dst];
    const float4 xv = *(const float4*)(xw + (size_t)src * C_DIM + lane * 4);
    float* ap = agg + (size_t)dst * C_DIM + lane * 4;
    atomicAdd(ap + 0, xv.x * w);
    atomicAdd(ap + 1, xv.y * w);
    atomicAdd(ap + 2, xv.z * w);
    atomicAdd(ap + 3, xv.w * w);
}

// ---------------- WMMA GEMM: out = act(X[bf16] @ W[bf16]^T + bias) ----------------
// X: nrows x 128, W: ncols x 128 (row-major, ncols<=128). 128 rows per block,
// 8 waves each owning a 16-row strip across 8 col-tiles. Tiles staged in LDS
// via async global->LDS B128 copies (ASYNCcnt). OOB rows/cols are simply not
// loaded: stale LDS only affects D rows/cols that are never stored.

__global__ __launch_bounds__(256, 1) void k_gemm128(
    const unsigned short* __restrict__ Xbf, int nrows,
    const unsigned short* __restrict__ Wbf, const float* __restrict__ bias,
    int ncols, int act, float* __restrict__ outF, int ldo,
    unsigned short* __restrict__ outBf)
{
    __shared__ unsigned short Wl[C_DIM * C_DIM];
    __shared__ unsigned short Xl[C_DIM * C_DIM];
    const int tid = threadIdx.x;
    const int row0 = blockIdx.x * 128;

    if (ncols == C_DIM) {
        #pragma unroll
        for (int it = 0; it < 8; ++it) {
            int chunk = tid + 256 * it;                       // 2048 x 16B = 32KB
            async_g2l_b128(&Wl[chunk * 8], (const char*)Wbf + (size_t)chunk * 16);
        }
    } else {
        for (int i = tid; i < C_DIM * C_DIM; i += 256) {
            int n = i >> 7;
            Wl[i] = (n < ncols) ? Wbf[i] : (unsigned short)0;
        }
    }
    #pragma unroll
    for (int it = 0; it < 8; ++it) {
        int chunk = tid + 256 * it;
        int row = row0 + (chunk >> 4);
        if (row < nrows)
            async_g2l_b128(&Xl[chunk * 8],
                           (const char*)Xbf + (size_t)row * 256 + (chunk & 15) * 16);
    }
    async_wait0();
    __syncthreads();

    const int wave = tid >> 5, lane = tid & 31;
    const int m16 = lane & 15, half = lane >> 4;

    v8f acc[8];
    #pragma unroll
    for (int c = 0; c < 8; ++c)
        #pragma unroll
        for (int r = 0; r < 8; ++r) acc[c][r] = 0.0f;

    #pragma unroll
    for (int t = 0; t < 4; ++t) {
        v16us au;
        #pragma unroll
        for (int e = 0; e < 16; ++e) {
            int k = 32 * t + e + ((e < 8) ? 0 : 8) + 8 * half;   // 16-bit A-layout K map
            au[e] = Xl[(16 * wave + m16) * C_DIM + k];
        }
        v16bf a = __builtin_bit_cast(v16bf, au);
        #pragma unroll
        for (int c = 0; c < 8; ++c) {
            v16us bu;
            #pragma unroll
            for (int e = 0; e < 16; ++e) {
                int k = 32 * t + e + ((e < 8) ? 0 : 8) + 8 * half;
                bu[e] = Wl[(16 * c + m16) * C_DIM + k];          // B[k][n] = W[n][k]
            }
            v16bf b = __builtin_bit_cast(v16bf, bu);
            acc[c] = __builtin_amdgcn_wmma_f32_16x16x32_bf16(
                false, a, false, b, (short)0, acc[c], false, false);
        }
    }

    #pragma unroll
    for (int c = 0; c < 8; ++c) {
        int n = 16 * c + m16;
        if (n >= ncols) continue;
        float bv = bias ? bias[n] : 0.0f;
        #pragma unroll
        for (int r = 0; r < 8; ++r) {
            int m = 16 * wave + r + 8 * half;                    // f32 C/D layout
            int row = row0 + m;
            if (row >= nrows) continue;
            float v = acc[c][r] + bv;
            if (act == 1) v = fmaxf(v, 0.0f);
            else if (act == 2) v = 1.0f / (1.0f + __expf(-v));
            if (outF)  outF[(size_t)row * ldo + n] = v;
            if (outBf) outBf[(size_t)row * C_DIM + n] = f2bf(v);
        }
    }
}

// ---------------- WMMA kv-reduction: kvT[m][h] += sum_n K[n][h] * V[n][m] ----------------
// K and V chunks are stored channel-major in LDS (padded stride 40 ushorts) so
// both A and B fragments are two contiguous 16B runs -> ds_load_b128 pairs.

#define KV_CHUNK 1024
#define KV_STRIDE 40
__global__ __launch_bounds__(256, 1) void k_kv_reduce(
    const unsigned short* __restrict__ Kbf, const unsigned short* __restrict__ Vbf,
    int nrows, float* __restrict__ kvT)
{
    __shared__ unsigned short KlT[C_DIM * KV_STRIDE];   // [channel][local row]
    __shared__ unsigned short VlT[C_DIM * KV_STRIDE];
    const int tid = threadIdx.x;
    const int base = blockIdx.x * KV_CHUNK;
    const int wave = tid >> 5, lane = tid & 31;
    const int m16 = lane & 15, half = lane >> 4;

    v8f acc[8];
    #pragma unroll
    for (int c = 0; c < 8; ++c)
        #pragma unroll
        for (int r = 0; r < 8; ++r) acc[c][r] = 0.0f;

    for (int s = 0; s < KV_CHUNK; s += 32) {
        for (int i = tid; i < 32 * C_DIM; i += 256) {
            int n = i >> 7;                              // local row 0..31
            int c = i & 127;                             // channel
            int r = base + s + n;
            bool ok = (r < nrows);
            unsigned short kw = ok ? Kbf[(size_t)r * C_DIM + c] : (unsigned short)0;
            unsigned short vw = ok ? Vbf[(size_t)r * C_DIM + c] : (unsigned short)0;
            KlT[c * KV_STRIDE + n] = kw;
            VlT[c * KV_STRIDE + n] = vw;
        }
        __syncthreads();
        v16us au;
        #pragma unroll
        for (int e = 0; e < 16; ++e) {
            int k = e + ((e < 8) ? 0 : 8) + 8 * half;    // local row index 0..31
            au[e] = KlT[(16 * wave + m16) * KV_STRIDE + k];   // A[h][n'] = K[n'][h]
        }
        v16bf a = __builtin_bit_cast(v16bf, au);
        #pragma unroll
        for (int c = 0; c < 8; ++c) {
            v16us bu;
            #pragma unroll
            for (int e = 0; e < 16; ++e) {
                int k = e + ((e < 8) ? 0 : 8) + 8 * half;
                bu[e] = VlT[(16 * c + m16) * KV_STRIDE + k];  // B[n'][m] = V[n'][m]
            }
            v16bf b = __builtin_bit_cast(v16bf, bu);
            acc[c] = __builtin_amdgcn_wmma_f32_16x16x32_bf16(
                false, a, false, b, (short)0, acc[c], false, false);
        }
        __syncthreads();
    }

    #pragma unroll
    for (int c = 0; c < 8; ++c) {
        int m = 16 * c + m16;                                    // v channel
        #pragma unroll
        for (int r = 0; r < 8; ++r) {
            int h = 16 * wave + r + 8 * half;                    // k channel
            atomicAdd(&kvT[(size_t)m * C_DIM + h], acc[c][r]);
        }
    }
}

__global__ void k_colsum(const float* __restrict__ kf, int nrows, float* __restrict__ ksum) {
    int c = threadIdx.x;                 // 128 threads
    int r0 = blockIdx.x * 512;
    int r1 = r0 + 512; if (r1 > nrows) r1 = nrows;
    float s = 0.0f;
    for (int r = r0; r < r1; ++r) s += kf[(size_t)r * C_DIM + c];
    atomicAdd(&ksum[c], s);
}

// ---------------- fused local-layer epilogue (wave per row) ----------------
// xv = BN(agg + xl); relu; t = h*xv; x = 0.1*LN(t) + 0.9*xv; xloc += x; xbf = bf16(x)
__global__ __launch_bounds__(256) void k_local_post(
    const float* __restrict__ agg, const float* __restrict__ xl, const float* __restrict__ h,
    const float* __restrict__ rm, const float* __restrict__ rv,
    const float* __restrict__ bng, const float* __restrict__ bnb,
    const float* __restrict__ lng, const float* __restrict__ lnb,
    float* __restrict__ xloc, unsigned short* __restrict__ xbf, int nrows)
{
    int wave = threadIdx.x >> 5, lane = threadIdx.x & 31;
    int row = blockIdx.x * 8 + wave;
    if (row >= nrows) return;
    int c0 = lane * 4;
    size_t off = (size_t)row * C_DIM + c0;
    float4 av = *(const float4*)(agg + off);
    float4 lv = *(const float4*)(xl + off);
    float4 hv = *(const float4*)(h + off);
    float4 rmv = *(const float4*)(rm + c0);
    float4 rvv = *(const float4*)(rv + c0);
    float4 bgv = *(const float4*)(bng + c0);
    float4 bbv = *(const float4*)(bnb + c0);
    float4 lgv = *(const float4*)(lng + c0);
    float4 lbv = *(const float4*)(lnb + c0);
    float xv[4], t[4];
    float a4[4] = {av.x, av.y, av.z, av.w}, l4[4] = {lv.x, lv.y, lv.z, lv.w};
    float h4[4] = {hv.x, hv.y, hv.z, hv.w};
    float rm4[4] = {rmv.x, rmv.y, rmv.z, rmv.w}, rv4[4] = {rvv.x, rvv.y, rvv.z, rvv.w};
    float bg4[4] = {bgv.x, bgv.y, bgv.z, bgv.w}, bb4[4] = {bbv.x, bbv.y, bbv.z, bbv.w};
    float lg4[4] = {lgv.x, lgv.y, lgv.z, lgv.w}, lb4[4] = {lbv.x, lbv.y, lbv.z, lbv.w};
    float lsum = 0.0f;
    #pragma unroll
    for (int j = 0; j < 4; ++j) {
        float v = a4[j] + l4[j];
        v = (v - rm4[j]) * rsqrtf(rv4[j] + 1e-5f) * bg4[j] + bb4[j];
        v = fmaxf(v, 0.0f);
        xv[j] = v;
        t[j] = h4[j] * v;
        lsum += t[j];
    }
    float mu = wred(lsum) * (1.0f / C_DIM);
    float q = 0.0f;
    #pragma unroll
    for (int j = 0; j < 4; ++j) { float d = t[j] - mu; q += d * d; }
    float rstd = rsqrtf(wred(q) * (1.0f / C_DIM) + 1e-5f);
    #pragma unroll
    for (int j = 0; j < 4; ++j) {
        float ln = (t[j] - mu) * rstd * lg4[j] + lb4[j];
        float xn = (1.0f - BETA) * ln + BETA * xv[j];
        xloc[off + j] += xn;
        xbf[off + j] = f2bf(xn);
    }
}

// plain LN over rows -> bf16 (for xg = LN(x_local))
__global__ __launch_bounds__(256) void k_lnrows(
    const float* __restrict__ src, const float* __restrict__ g, const float* __restrict__ b,
    unsigned short* __restrict__ outBf, int nrows)
{
    int wave = threadIdx.x >> 5, lane = threadIdx.x & 31;
    int row = blockIdx.x * 8 + wave;
    if (row >= nrows) return;
    int c0 = lane * 4;
    size_t off = (size_t)row * C_DIM + c0;
    float4 sv = *(const float4*)(src + off);
    float4 gv = *(const float4*)(g + c0);
    float4 bv = *(const float4*)(b + c0);
    float s4[4] = {sv.x, sv.y, sv.z, sv.w};
    float g4[4] = {gv.x, gv.y, gv.z, gv.w}, b4[4] = {bv.x, bv.y, bv.z, bv.w};
    float lsum = s4[0] + s4[1] + s4[2] + s4[3];
    float mu = wred(lsum) * (1.0f / C_DIM);
    float q = 0.0f;
    #pragma unroll
    for (int j = 0; j < 4; ++j) { float d = s4[j] - mu; q += d * d; }
    float rstd = rsqrtf(wred(q) * (1.0f / C_DIM) + 1e-5f);
    #pragma unroll
    for (int j = 0; j < 4; ++j)
        outBf[off + j] = f2bf((s4[j] - mu) * rstd * g4[j] + b4[j]);
}

// global-attn mid: attn = num/(q.ksum+1e-6); out = LN(attn)*(h+BETA) -> bf16
__global__ __launch_bounds__(256) void k_attn_mid(
    const float* __restrict__ num, const float* __restrict__ kf,
    const float* __restrict__ ksum, const float* __restrict__ h,
    const float* __restrict__ lng, const float* __restrict__ lnb,
    unsigned short* __restrict__ midBf, int nrows)
{
    int wave = threadIdx.x >> 5, lane = threadIdx.x & 31;
    int row = blockIdx.x * 8 + wave;
    if (row >= nrows) return;
    int c0 = lane * 4;
    size_t off = (size_t)row * C_DIM + c0;
    float4 nv = *(const float4*)(num + off);
    float4 kv = *(const float4*)(kf + off);
    float4 sv = *(const float4*)(ksum + c0);
    float4 hv = *(const float4*)(h + off);
    float4 gv = *(const float4*)(lng + c0);
    float4 bv = *(const float4*)(lnb + c0);
    float n4[4] = {nv.x, nv.y, nv.z, nv.w}, k4[4] = {kv.x, kv.y, kv.z, kv.w};
    float s4[4] = {sv.x, sv.y, sv.z, sv.w}, h4[4] = {hv.x, hv.y, hv.z, hv.w};
    float g4[4] = {gv.x, gv.y, gv.z, gv.w}, b4[4] = {bv.x, bv.y, bv.z, bv.w};
    float dp = k4[0] * s4[0] + k4[1] * s4[1] + k4[2] * s4[2] + k4[3] * s4[3];
    float den = wred(dp) + 1e-6f;
    float at[4], lsum = 0.0f;
    #pragma unroll
    for (int j = 0; j < 4; ++j) { at[j] = n4[j] / den; lsum += at[j]; }
    float mu = wred(lsum) * (1.0f / C_DIM);
    float q = 0.0f;
    #pragma unroll
    for (int j = 0; j < 4; ++j) { float d = at[j] - mu; q += d * d; }
    float rstd = rsqrtf(wred(q) * (1.0f / C_DIM) + 1e-5f);
    #pragma unroll
    for (int j = 0; j < 4; ++j) {
        float ln = (at[j] - mu) * rstd * g4[j] + b4[j];
        midBf[off + j] = f2bf(ln * (h4[j] + BETA));
    }
}

// ---------------------------------------------------------------------------

extern "C" void kernel_launch(void* const* d_in, const int* in_sizes, int n_in,
                              void* d_out, int out_size, void* d_ws, size_t ws_size,
                              hipStream_t stream) {
    const int L = 7, G = 2, O = 40;
    const int N = in_sizes[0] / C_DIM;
    const int E = in_sizes[1] / 2;

    const float* x_in  = (const float*)d_in[0];
    const int*   eidx  = (const int*)d_in[1];
    const float* Wh    = (const float*)d_in[2];
    const float* bh    = (const float*)d_in[3];
    const float* Wg    = (const float*)d_in[4];
    const float* bg    = (const float*)d_in[5];
    const float* Wlw   = (const float*)d_in[6];
    const float* bl    = (const float*)d_in[7];
    const float* ln_g  = (const float*)d_in[8];
    const float* ln_b  = (const float*)d_in[9];
    const float* bn_g  = (const float*)d_in[10];
    const float* bn_b  = (const float*)d_in[11];
    const float* bn_rm = (const float*)d_in[12];
    const float* bn_rv = (const float*)d_in[13];
    const float* gWh   = (const float*)d_in[14];
    const float* gbh   = (const float*)d_in[15];
    const float* gWk   = (const float*)d_in[16];
    const float* gWv   = (const float*)d_in[17];
    const float* gln_g = (const float*)d_in[18];
    const float* gln_b = (const float*)d_in[19];
    const float* gWo   = (const float*)d_in[20];
    const float* gbo   = (const float*)d_in[21];
    const float* fln_g = (const float*)d_in[22];
    const float* fln_b = (const float*)d_in[23];
    const float* Wp    = (const float*)d_in[24];
    const float* bp    = (const float*)d_in[25];
    float* out = (float*)d_out;

    // ---- workspace carve ----
    const size_t NC = (size_t)N * C_DIM;
    float* hb    = (float*)d_ws;                 // N x 128
    float* xwb   = hb + NC;                      // xw / k(f32)
    float* aggb  = xwb + NC;                     // agg / num
    float* xlb   = aggb + NC;                    // xl
    float* xlocb = xlb + NC;                     // x_local
    unsigned short* xbf = (unsigned short*)(xlocb + NC);   // x / xg in bf16
    unsigned short* kbf = xbf + NC;
    unsigned short* vbf = kbf + NC;              // also reused as mid-bf16
    float* deg   = (float*)(vbf + NC);
    float* dinv  = deg + N;
    float* kvT   = dinv + N;                     // 128x128
    float* ksum  = kvT + C_DIM * C_DIM;          // 128
    unsigned short* kvTbf = (unsigned short*)(ksum + C_DIM);
    unsigned short* wWh  = kvTbf + C_DIM * C_DIM;
    unsigned short* wWg  = wWh  + (size_t)L * C_DIM * C_DIM;
    unsigned short* wWl  = wWg  + (size_t)L * C_DIM * C_DIM;
    unsigned short* wgWh = wWl  + (size_t)L * C_DIM * C_DIM;
    unsigned short* wgWk = wgWh + (size_t)G * C_DIM * C_DIM;
    unsigned short* wgWv = wgWk + (size_t)G * C_DIM * C_DIM;
    unsigned short* wgWo = wgWv + (size_t)G * C_DIM * C_DIM;
    unsigned short* wWp  = wgWo + (size_t)G * C_DIM * C_DIM;   // 40x128

    const int gemmGrid = (N + 127) / 128;
    const int rowGrid  = (N + 7) / 8;
    (void)ws_size; (void)n_in; (void)out_size;

    // ---- weight conversion to bf16 ----
    k_cvt_bf16<<<512, 256, 0, stream>>>(Wh,  wWh,  (size_t)L * C_DIM * C_DIM);
    k_cvt_bf16<<<512, 256, 0, stream>>>(Wg,  wWg,  (size_t)L * C_DIM * C_DIM);
    k_cvt_bf16<<<512, 256, 0, stream>>>(Wlw, wWl,  (size_t)L * C_DIM * C_DIM);
    k_cvt_bf16<<<128, 256, 0, stream>>>(gWh, wgWh, (size_t)G * C_DIM * C_DIM);
    k_cvt_bf16<<<128, 256, 0, stream>>>(gWk, wgWk, (size_t)G * C_DIM * C_DIM);
    k_cvt_bf16<<<128, 256, 0, stream>>>(gWv, wgWv, (size_t)G * C_DIM * C_DIM);
    k_cvt_bf16<<<128, 256, 0, stream>>>(gWo, wgWo, (size_t)G * C_DIM * C_DIM);
    k_cvt_bf16<<<32,  256, 0, stream>>>(Wp,  wWp,  (size_t)O * C_DIM);

    // ---- x -> bf16; zero accumulators ----
    k_cvt_bf16<<<2048, 256, 0, stream>>>(x_in, xbf, NC);
    k_zero_f32<<<2048, 256, 0, stream>>>(xlocb, NC);
    k_zero_f32<<<256,  256, 0, stream>>>(deg, (size_t)N);

    // ---- degree / dinv ----
    k_deg<<<2048, 256, 0, stream>>>(eidx, E, deg);
    k_dinv<<<(N + 255) / 256, 256, 0, stream>>>(deg, dinv, N);

    // ---- local GCN stack ----
    for (int i = 0; i < L; ++i) {
        const size_t wo = (size_t)i * C_DIM * C_DIM;
        k_gemm128<<<gemmGrid, 256, 0, stream>>>(xbf, N, wWh + wo, bh + i * C_DIM,
                                                C_DIM, /*relu*/1, hb, C_DIM, nullptr);
        k_gemm128<<<gemmGrid, 256, 0, stream>>>(xbf, N, wWg + wo, nullptr,
                                                C_DIM, 0, xwb, C_DIM, nullptr);
        k_gemm128<<<gemmGrid, 256, 0, stream>>>(xbf, N, wWl + wo, bl + i * C_DIM,
                                                C_DIM, 0, xlb, C_DIM, nullptr);
        k_scatter_init<<<2048, 256, 0, stream>>>(xwb, dinv, bg + i * C_DIM, aggb, NC);
        k_edge_scatter<<<(E + 7) / 8, 256, 0, stream>>>(eidx, E, dinv, xwb, aggb);
        k_local_post<<<rowGrid, 256, 0, stream>>>(
            aggb, xlb, hb,
            bn_rm + i * C_DIM, bn_rv + i * C_DIM, bn_g + i * C_DIM, bn_b + i * C_DIM,
            ln_g + i * C_DIM, ln_b + i * C_DIM, xlocb, xbf, N);
    }

    // ---- xg = LN(x_local) ----
    k_lnrows<<<rowGrid, 256, 0, stream>>>(xlocb, fln_g, fln_b, xbf, N);

    // ---- global linear attention ----
    for (int j = 0; j < G; ++j) {
        const size_t wo = (size_t)j * C_DIM * C_DIM;
        k_gemm128<<<gemmGrid, 256, 0, stream>>>(xbf, N, wgWh + wo, gbh + j * C_DIM,
                                                C_DIM, 0, hb, C_DIM, nullptr);
        k_gemm128<<<gemmGrid, 256, 0, stream>>>(xbf, N, wgWk + wo, nullptr,
                                                C_DIM, /*sigmoid*/2, xwb, C_DIM, kbf);
        k_gemm128<<<gemmGrid, 256, 0, stream>>>(xbf, N, wgWv + wo, nullptr,
                                                C_DIM, 0, nullptr, C_DIM, vbf);
        k_zero_f32<<<17, 256, 0, stream>>>(kvT, (size_t)(C_DIM * C_DIM + C_DIM));
        k_kv_reduce<<<(N + KV_CHUNK - 1) / KV_CHUNK, 256, 0, stream>>>(kbf, vbf, N, kvT);
        k_colsum<<<(N + 511) / 512, 128, 0, stream>>>(xwb, N, ksum);
        k_cvt_bf16<<<32, 256, 0, stream>>>(kvT, kvTbf, (size_t)C_DIM * C_DIM);
        // num = q @ kv  (W = kv^T so X@W^T == q@kv)
        k_gemm128<<<gemmGrid, 256, 0, stream>>>(kbf, N, kvTbf, nullptr,
                                                C_DIM, 0, aggb, C_DIM, nullptr);
        k_attn_mid<<<rowGrid, 256, 0, stream>>>(aggb, xwb, ksum, hb,
                                                gln_g + j * C_DIM, gln_b + j * C_DIM,
                                                vbf, N);
        k_gemm128<<<gemmGrid, 256, 0, stream>>>(vbf, N, wgWo + wo, gbo + j * C_DIM,
                                                C_DIM, /*relu*/1, nullptr, C_DIM, xbf);
    }

    // ---- final projection to O=40 ----
    k_gemm128<<<gemmGrid, 256, 0, stream>>>(xbf, N, wWp, bp, O, 0, out, O, nullptr);
}
